// NUSpectralConv2d_19971597926544
// MI455X (gfx1250) — compile-verified, matrix operand order checked by reading, not run on hardware
//
#include <hip/hip_runtime.h>
#include <hip/hip_bf16.h>
#include <cstdint>

typedef __attribute__((ext_vector_type(16))) _Float16 v16h;
typedef __attribute__((ext_vector_type(8)))  _Float16 v8h;
typedef __attribute__((ext_vector_type(8)))  float    v8f;

#define BB 4
#define CC 32
#define NN 8192
#define SS 8
#define PP 1024
#define MM 12
#define XX 25
#define KH 13
#define NJ 626          // 600 (i1 re/im) + 26 (i2 re/im); padded to 640
#define EPSF 1e-6f
#define PIF 3.14159265358979f
#define NORMF 0.04f     // 1/sqrt(25*25)

// ---- workspace layout (float offsets) ----
// xg stored as f16 (half the region used), e-tables interleaved [p][52]:
//   [0..12]=cos1, [13..25]=sin1, [26..38]=cos2, [39..51]=sin2
#define OFF_XG  0L
#define OFF_E   (OFF_XG  + (long)BB*SS*CC*PP)
#define OFF_XMN (OFF_E   + (long)BB*SS*PP*52)
#define OFF_XMX (OFF_XMN + (long)BB*SS*CC)
#define OFF_C1  (OFF_XMX + (long)BB*SS*CC)
#define OFF_C2  (OFF_C1  + (long)BB*SS*CC*600)
#define OFF_D1  (OFF_C2  + (long)BB*SS*CC*26)
#define OFF_D2  (OFF_D1  + (long)BB*CC*600)
#define OFF_F1  (OFF_D2  + (long)BB*CC*26)
#define OFF_F2  (OFF_F1  + (long)BB*CC*600)
#define OFF_GH  (OFF_F2  + (long)BB*CC*26)          // f16 [bs][c][640] padded
#define OFF_OR  (OFF_GH  + (long)BB*SS*CC*320)      // 640 halves = 320 floats

__device__ __forceinline__ v8f wmma_f32_f16(v16h a, v16h b, v8f c) {
  return __builtin_amdgcn_wmma_f32_16x16x32_f16(false, a, false, b, (short)0, c,
                                                false, false);
}

// CDNA5 async global->LDS copy (ASYNCcnt) -- 16B granule, both sides 16B aligned
__device__ __forceinline__ void async_copy_b128(void* lds, const void* g) {
  unsigned l = (unsigned)(uintptr_t)lds;                 // low 32b of generic = LDS offset
  unsigned long long a = (unsigned long long)(uintptr_t)g;
  asm volatile("global_load_async_to_lds_b128 %0, %1, off"
               :: "v"(l), "v"(a) : "memory");
}
__device__ __forceinline__ void async_wait0() {
  asm volatile("s_wait_asynccnt 0x0" ::: "memory");
}

// mode-pair decode: pr<300 -> i1 (x=pr/12,y=pr%12); 300..312 -> i2; >=313 zero
__device__ __forceinline__ int pair_code(int pr) {
  if (pr >= 313) return 1 << 12;
  if (pr < 300) {
    int xr = pr / 12, yr = pr % 12;
    int kx = xr - 12;
    int ax = kx < 0 ? -kx : kx;
    int sn = kx > 0 ? 1 : 0;
    int ay = 12 - yr;
    return ax | (ay << 5) | (sn << 10);
  }
  return (12 - (pr - 300)) | (1 << 11);
}

// ============== K1: gather, om normalization, sincos tables, xg min/max =====
__global__ void __launch_bounds__(256)
k1_prep(const float* __restrict__ x, const float* __restrict__ loc,
        const int* __restrict__ ind, float* __restrict__ ws) {
  __shared__ int   idx[PP];
  __shared__ float l0s[PP], l1s[PP];
  __shared__ float red[256];
  const int bs = blockIdx.x;
  const int b = bs / SS, s = bs % SS;
  const int t = threadIdx.x;

  for (int p = t; p < PP; p += 256) {
    int ii = ind[b * NN + s * PP + p];
    idx[p] = ii;
    l0s[p] = loc[((long)b * NN + ii) * 2 + 0];
    l1s[p] = loc[((long)b * NN + ii) * 2 + 1];
  }
  __syncthreads();

  float mn0 = 1e30f, mx0 = -1e30f, mn1 = 1e30f, mx1 = -1e30f;
  for (int p = t; p < PP; p += 256) {
    mn0 = fminf(mn0, l0s[p]); mx0 = fmaxf(mx0, l0s[p]);
    mn1 = fminf(mn1, l1s[p]); mx1 = fmaxf(mx1, l1s[p]);
  }
  auto bred = [&](float v, bool ismax) -> float {
    red[t] = v; __syncthreads();
    for (int o = 128; o > 0; o >>= 1) {
      if (t < o) red[t] = ismax ? fmaxf(red[t], red[t + o]) : fminf(red[t], red[t + o]);
      __syncthreads();
    }
    float r = red[0]; __syncthreads(); return r;
  };
  mn0 = bred(mn0, false); mx0 = bred(mx0, true);
  mn1 = bred(mn1, false); mx1 = bred(mx1, true);

  const float i0 = 1.0f / (mx0 - mn0 + EPSF);
  const float i1 = 1.0f / (mx1 - mn1 + EPSF);
  for (int p = t; p < PP; p += 256) {
    float om1 = (l0s[p] - mn0) * i0 * (2.0f * PIF) - PIF;
    float om2 = (l1s[p] - mn1) * i1 * (2.0f * PIF) - PIF;
    long base = OFF_E + ((long)bs * PP + p) * 52;
    for (int k = 0; k < KH; k++) {
      float sv, cv;
      __sincosf(om1 * (float)k, &sv, &cv);
      ws[base + k] = cv; ws[base + 13 + k] = sv;
      __sincosf(om2 * (float)k, &sv, &cv);
      ws[base + 26 + k] = cv; ws[base + 39 + k] = sv;
    }
  }
  // one wave per 4 channels: gather xg (stored f16) + f32 min/max (wave32)
  _Float16* xgh = (_Float16*)(ws + OFF_XG);
  const int w = t >> 5, lane = t & 31;
  for (int c = w * 4; c < w * 4 + 4; c++) {
    float mn = 1e30f, mx = -1e30f;
    for (int i = 0; i < 32; i++) {
      int p = i * 32 + lane;
      float v = x[((long)b * CC + c) * NN + idx[p]];
      xgh[((long)bs * CC + c) * PP + p] = (_Float16)v;
      mn = fminf(mn, v); mx = fmaxf(mx, v);
    }
    for (int o = 16; o > 0; o >>= 1) {
      mn = fminf(mn, __shfl_xor(mn, o));
      mx = fmaxf(mx, __shfl_xor(mx, o));
    }
    if (lane == 0) {
      ws[OFF_XMN + bs * CC + c] = mn;
      ws[OFF_XMX + bs * CC + c] = mx;
    }
  }
}

// ============== K2: forward NUDFT WMMA GEMM: coef[c,j] = sum_p xg * W =======
// grid = 32 (b,s) * 5 j-chunks of 128.  Basis W built per 32-p tile in LDS.
__global__ void __launch_bounds__(256)
k2_forward(float* __restrict__ ws) {
  __shared__ __align__(16) float    es[32 * 52];     // staged e-tables
  __shared__ __align__(16) _Float16 Wsh[128 * 32];   // [j][k] K-contiguous
  __shared__ int dtab[320];
  const int bs = blockIdx.x / 5;
  const int jc = blockIdx.x % 5;
  const int t = threadIdx.x, w = t >> 5, lane = t & 31;
  const int mt = w >> 2, nt0 = (w * 2) & 7;
  const int lid = lane & 15, kb = (lane & 16) ? 8 : 0;
  v8f acc0 = {0, 0, 0, 0, 0, 0, 0, 0};
  v8f acc1 = {0, 0, 0, 0, 0, 0, 0, 0};
  const int crow = mt * 16 + lid;
  const _Float16* xgh = (const _Float16*)(ws + OFF_XG);
  const long abase = ((long)bs * CC + crow) * PP + kb;
  const int q = t & 63, kk0 = (t >> 6) * 8;

  for (int i = t; i < 320; i += 256) dtab[i] = pair_code(i);

  for (int kt = 0; kt < 32; kt++) {
    const int p0 = kt * 32;
    __syncthreads();
    {  // async-stage 32x52 floats of e-tables (contiguous, 16B aligned)
      const char* gsrc = (const char*)(ws + OFF_E + ((long)bs * PP + p0) * 52);
      for (int i = t; i < 416; i += 256)
        async_copy_b128(&es[i * 4], gsrc + i * 16);
      if (kt + 1 < 32)
        __builtin_prefetch((const void*)(ws + OFF_E + ((long)bs * PP + p0 + 32) * 52), 0, 1);
      async_wait0();
    }
    __syncthreads();
    {  // build W: each thread = one mode pair, 8-wide K run, vector LDS stores
      int code = dtab[jc * 64 + q];
      int ax = code & 31, ay = (code >> 5) & 31;
      int sn = (code >> 10) & 1, is2 = (code >> 11) & 1, isz = (code >> 12) & 1;
      v8h wre, wim;
      #pragma unroll
      for (int u = 0; u < 8; u++) {
        const float* ep = es + (kk0 + u) * 52;
        float c1v = ep[ax], s1v = ep[13 + ax];
        float wr, wi;
        if (isz) { wr = 0.f; wi = 0.f; }
        else if (is2) { wr = c1v * NORMF; wi = s1v * NORMF; }
        else {
          float e1i = sn ? -s1v : s1v;
          float c2v = ep[26 + ay], s2v = ep[39 + ay];
          wr = (c1v * c2v - e1i * s2v) * NORMF;
          wi = (c1v * s2v + e1i * c2v) * NORMF;
        }
        wre[u] = (_Float16)wr; wim[u] = (_Float16)wi;
      }
      *(v8h*)(Wsh + (2 * q) * 32 + kk0)     = wre;
      *(v8h*)(Wsh + (2 * q + 1) * 32 + kk0) = wim;
    }
    __syncthreads();
    // fragments: 2x b128 each
    v8h alo = *(const v8h*)(xgh + abase + p0);
    v8h ahi = *(const v8h*)(xgh + abase + p0 + 16);
    v16h af = __builtin_shufflevector(alo, ahi, 0,1,2,3,4,5,6,7,8,9,10,11,12,13,14,15);
    const _Float16* wrow0 = Wsh + (nt0 * 16 + lid) * 32 + kb;
    const _Float16* wrow1 = wrow0 + 16 * 32;
    v8h b0l = *(const v8h*)(wrow0), b0h = *(const v8h*)(wrow0 + 16);
    v8h b1l = *(const v8h*)(wrow1), b1h = *(const v8h*)(wrow1 + 16);
    v16h bf0 = __builtin_shufflevector(b0l, b0h, 0,1,2,3,4,5,6,7,8,9,10,11,12,13,14,15);
    v16h bf1 = __builtin_shufflevector(b1l, b1h, 0,1,2,3,4,5,6,7,8,9,10,11,12,13,14,15);
    acc0 = wmma_f32_f16(af, bf0, acc0);
    acc1 = wmma_f32_f16(af, bf1, acc1);
  }
  for (int r = 0; r < 8; r++) {
    int Mr = mt * 16 + ((lane & 16) ? (r + 8) : r);
    int j0 = jc * 128 + nt0 * 16 + lid;
    int j1 = j0 + 16;
    long base = (long)bs * CC + Mr;
    float v0 = acc0[r], v1 = acc1[r];
    if (j0 < 600)      ws[OFF_C1 + base * 600 + j0] = v0;
    else if (j0 < NJ)  ws[OFF_C2 + base * 26 + (j0 - 600)] = v0;
    if (j1 < 600)      ws[OFF_C1 + base * 600 + j1] = v1;
    else if (j1 < NJ)  ws[OFF_C2 + base * 26 + (j1 - 600)] = v1;
  }
}

// ============== K3: tiny complex mixing einsums (VALU) ======================
__global__ void k3_smix(const float* __restrict__ w1s_re, const float* __restrict__ w1s_im,
                        const float* __restrict__ w2s_re, const float* __restrict__ w2s_im,
                        float* __restrict__ ws) {
  int tid = blockIdx.x * 256 + threadIdx.x;
  if (tid >= BB * CC * 313) return;
  int pr = tid % 313, bi = tid / 313;
  int b = bi >> 5, ci = bi & 31;
  float dre = 0.f, dim_ = 0.f;
  if (pr < 300) {
    int xr = pr / MM, yr = pr % MM;
    for (int s = 0; s < SS; s++) {
      long cb = OFF_C1 + (((long)(b * SS + s) * CC + ci) * 600 + pr * 2);
      float cr = ws[cb], cim = ws[cb + 1];
      float wr = w1s_re[(s * XX + xr) * MM + yr], wi = w1s_im[(s * XX + xr) * MM + yr];
      dre += cr * wr - cim * wi; dim_ += cr * wi + cim * wr;
    }
    ws[OFF_D1 + (long)bi * 600 + pr * 2]     = dre;
    ws[OFF_D1 + (long)bi * 600 + pr * 2 + 1] = dim_;
  } else {
    int xp = pr - 300;
    for (int s = 0; s < SS; s++) {
      long cb = OFF_C2 + (((long)(b * SS + s) * CC + ci) * 26 + xp * 2);
      float cr = ws[cb], cim = ws[cb + 1];
      float wr = w2s_re[s * KH + xp], wi = w2s_im[s * KH + xp];
      dre += cr * wr - cim * wi; dim_ += cr * wi + cim * wr;
    }
    ws[OFF_D2 + (long)bi * 26 + xp * 2]     = dre;
    ws[OFF_D2 + (long)bi * 26 + xp * 2 + 1] = dim_;
  }
}

__global__ void k3_cmix(const float* __restrict__ w1_re, const float* __restrict__ w1_im,
                        const float* __restrict__ w2_re, const float* __restrict__ w2_im,
                        float* __restrict__ ws) {
  int tid = blockIdx.x * 256 + threadIdx.x;
  if (tid >= BB * CC * 313) return;
  int pr = tid % 313, bo = tid / 313;
  int b = bo >> 5, o = bo & 31;
  float fre = 0.f, fim = 0.f;
  if (pr < 300) {
    int xr = pr / MM, yr = pr % MM;
    for (int i = 0; i < CC; i++) {
      long db = OFF_D1 + ((long)(b * CC + i) * 600 + pr * 2);
      float dr = ws[db], di = ws[db + 1];
      long wb = ((long)(i * CC + o) * XX + xr) * MM + yr;
      float wr = w1_re[wb], wi = w1_im[wb];
      fre += dr * wr - di * wi; fim += dr * wi + di * wr;
    }
    ws[OFF_F1 + (long)bo * 600 + pr * 2]     = fre;
    ws[OFF_F1 + (long)bo * 600 + pr * 2 + 1] = fim;
  } else {
    int xp = pr - 300;
    for (int i = 0; i < CC; i++) {
      long db = OFF_D2 + ((long)(b * CC + i) * 26 + xp * 2);
      float dr = ws[db], di = ws[db + 1];
      float wr = w2_re[(i * CC + o) * KH + xp], wi = w2_im[(i * CC + o) * KH + xp];
      fre += dr * wr - di * wi; fim += dr * wi + di * wr;
    }
    ws[OFF_F2 + (long)bo * 26 + xp * 2]     = fre;
    ws[OFF_F2 + (long)bo * 26 + xp * 2 + 1] = fim;
  }
}

// s-expand + pack G into padded f16 [bs][c][640] (zeros in 626..639)
__global__ void k3_sexp(const float* __restrict__ w1sb_re, const float* __restrict__ w1sb_im,
                        const float* __restrict__ w2sb_re, const float* __restrict__ w2sb_im,
                        float* __restrict__ ws) {
  int tid = blockIdx.x * 256 + threadIdx.x;
  if (tid >= BB * CC * SS * 320) return;
  int pr = tid % 320; int r = tid / 320;
  int s = r & 7; r >>= 3; int o = r & 31; int b = r >> 5;
  _Float16* gh = (_Float16*)(ws + OFF_GH);
  long grow = ((long)(b * SS + s) * CC + o) * 640;
  if (pr >= 313) {
    int j = 626 + 2 * (pr - 313);
    gh[grow + j] = (_Float16)0.f; gh[grow + j + 1] = (_Float16)0.f;
    return;
  }
  long fb; float wr, wi; int j;
  if (pr < 300) {
    int xr = pr / MM, yr = pr % MM;
    fb = OFF_F1 + (long)(b * CC + o) * 600 + pr * 2;
    wr = w1sb_re[(s * XX + xr) * MM + yr]; wi = w1sb_im[(s * XX + xr) * MM + yr];
    j = 2 * pr;
  } else {
    int xp = pr - 300;
    fb = OFF_F2 + (long)(b * CC + o) * 26 + xp * 2;
    wr = w2sb_re[s * KH + xp]; wi = w2sb_im[s * KH + xp];
    j = 600 + 2 * xp;
  }
  float fr = ws[fb], fi = ws[fb + 1];
  gh[grow + j]     = (_Float16)(fr * wr - fi * wi);
  gh[grow + j + 1] = (_Float16)(fr * wi + fi * wr);
}

// ============== K4: inverse NUDFT (Hermitian-folded) WMMA GEMM ==============
// out[c,p] = sum_j G[c,j] * V[j,p].  grid = 32 (b,s) * 8 p-chunks of 128.
__global__ void __launch_bounds__(256)
k4_inverse(float* __restrict__ ws) {
  __shared__ __align__(16) float    es[128 * 52];    // 26KB staged e-tables
  __shared__ __align__(16) _Float16 Vsh[128 * 32];   // [p][j] K-contiguous
  __shared__ int dtab[320];
  const int bs = blockIdx.x >> 3;
  const int pc = blockIdx.x & 7;
  const int p0 = pc * 128;
  const int t = threadIdx.x, w = t >> 5, lane = t & 31;
  const int mt = w >> 2, nt0 = (w * 2) & 7;
  const int lid = lane & 15, kb = (lane & 16) ? 8 : 0;
  const int crow = mt * 16 + lid;
  const _Float16* gh = (const _Float16*)(ws + OFF_GH);
  const long gbase = ((long)bs * CC + crow) * 640 + kb;

  {  // async-stage 128x52 floats once (contiguous region, 16B aligned)
    const char* gsrc = (const char*)(ws + OFF_E + ((long)bs * PP + p0) * 52);
    for (int i = t; i < 1664; i += 256)
      async_copy_b128(&es[i * 4], gsrc + i * 16);
  }
  for (int i = t; i < 320; i += 256) dtab[i] = pair_code(i);
  async_wait0();
  __syncthreads();

  v8f acc0 = {0, 0, 0, 0, 0, 0, 0, 0};
  v8f acc1 = {0, 0, 0, 0, 0, 0, 0, 0};
  const int ppv = t & 127, gsel = t >> 7;
  const float* ep = es + ppv * 52;

  for (int kt = 0; kt < 20; kt++) {
    __syncthreads();
    {  // build V: thread = one p, 8 mode pairs, vector LDS stores
      v8h h0, h1;
      #pragma unroll
      for (int u = 0; u < 8; u++) {
        int code = dtab[kt * 16 + gsel * 8 + u];
        int ax = code & 31, ay = (code >> 5) & 31;
        int sn = (code >> 10) & 1, is2 = (code >> 11) & 1, isz = (code >> 12) & 1;
        float c1v = ep[ax], s1v = ep[13 + ax];
        float vr, vi;
        if (isz) { vr = 0.f; vi = 0.f; }
        else if (is2) {
          vr = (ax ? 2.f * NORMF : NORMF) * c1v;
          vi = 2.f * NORMF * s1v;              // s1[0]==0 at the DC term
        } else {
          float ai = sn ? s1v : -s1v;
          float cr = ep[26 + ay], ci = -ep[39 + ay];
          vr =  2.f * NORMF * (c1v * cr - ai * ci);
          vi = -2.f * NORMF * (c1v * ci + ai * cr);
        }
        if (u < 4) { h0[2*(u&3)] = (_Float16)vr; h0[2*(u&3)+1] = (_Float16)vi; }
        else       { h1[2*(u&3)] = (_Float16)vr; h1[2*(u&3)+1] = (_Float16)vi; }
      }
      *(v8h*)(Vsh + ppv * 32 + gsel * 16)     = h0;
      *(v8h*)(Vsh + ppv * 32 + gsel * 16 + 8) = h1;
    }
    __syncthreads();
    v8h alo = *(const v8h*)(gh + gbase + kt * 32);
    v8h ahi = *(const v8h*)(gh + gbase + kt * 32 + 16);
    v16h af = __builtin_shufflevector(alo, ahi, 0,1,2,3,4,5,6,7,8,9,10,11,12,13,14,15);
    const _Float16* vrow0 = Vsh + (nt0 * 16 + lid) * 32 + kb;
    const _Float16* vrow1 = vrow0 + 16 * 32;
    v8h b0l = *(const v8h*)(vrow0), b0h = *(const v8h*)(vrow0 + 16);
    v8h b1l = *(const v8h*)(vrow1), b1h = *(const v8h*)(vrow1 + 16);
    v16h bf0 = __builtin_shufflevector(b0l, b0h, 0,1,2,3,4,5,6,7,8,9,10,11,12,13,14,15);
    v16h bf1 = __builtin_shufflevector(b1l, b1h, 0,1,2,3,4,5,6,7,8,9,10,11,12,13,14,15);
    acc0 = wmma_f32_f16(af, bf0, acc0);
    acc1 = wmma_f32_f16(af, bf1, acc1);
  }
  for (int r = 0; r < 8; r++) {
    int Mr = mt * 16 + ((lane & 16) ? (r + 8) : r);
    int pa = p0 + nt0 * 16 + lid;
    ws[OFF_OR + ((long)bs * CC + Mr) * PP + pa]      = acc0[r];
    ws[OFF_OR + ((long)bs * CC + Mr) * PP + pa + 16] = acc1[r];
  }
}

// ============== K5: per-(b,c,s) renormalize + scatter by ind ================
__global__ void __launch_bounds__(256)
k5_finalize(const int* __restrict__ ind, const float* __restrict__ ws,
            float* __restrict__ out) {
  __shared__ float red[256];
  const int bsc = blockIdx.x;
  const int c = bsc & 31, bs = bsc >> 5;
  const int b = bs >> 3, s = bs & 7;
  const int t = threadIdx.x;
  float v[4]; float mn = 1e30f, mx = -1e30f;
  for (int q = 0; q < 4; q++) {
    v[q] = ws[OFF_OR + ((long)bs * CC + c) * PP + q * 256 + t];
    mn = fminf(mn, v[q]); mx = fmaxf(mx, v[q]);
  }
  red[t] = mn; __syncthreads();
  for (int o = 128; o > 0; o >>= 1) { if (t < o) red[t] = fminf(red[t], red[t + o]); __syncthreads(); }
  mn = red[0]; __syncthreads();
  red[t] = mx; __syncthreads();
  for (int o = 128; o > 0; o >>= 1) { if (t < o) red[t] = fmaxf(red[t], red[t + o]); __syncthreads(); }
  mx = red[0];
  float xmn = ws[OFF_XMN + bs * CC + c];
  float xmx = ws[OFF_XMX + bs * CC + c];
  float inv = 1.0f / (mx - mn + EPSF);
  float scl = xmx - xmn;
  for (int q = 0; q < 4; q++) {
    int p = q * 256 + t;
    float val = (v[q] - mn) * inv * scl + xmn;
    int n = ind[b * NN + s * PP + p];
    out[((long)b * CC + c) * NN + n] = val;
  }
}

extern "C" void kernel_launch(void* const* d_in, const int* in_sizes, int n_in,
                              void* d_out, int out_size, void* d_ws, size_t ws_size,
                              hipStream_t stream) {
  (void)in_sizes; (void)n_in; (void)out_size; (void)ws_size;
  const float* x       = (const float*)d_in[0];
  const float* loc     = (const float*)d_in[1];
  const int*   ind     = (const int*)d_in[2];
  const float* w1s_re  = (const float*)d_in[3];
  const float* w1s_im  = (const float*)d_in[4];
  const float* w2s_re  = (const float*)d_in[5];
  const float* w2s_im  = (const float*)d_in[6];
  const float* w1_re   = (const float*)d_in[7];
  const float* w1_im   = (const float*)d_in[8];
  const float* w2_re   = (const float*)d_in[9];
  const float* w2_im   = (const float*)d_in[10];
  const float* w1sb_re = (const float*)d_in[11];
  const float* w1sb_im = (const float*)d_in[12];
  const float* w2sb_re = (const float*)d_in[13];
  const float* w2sb_im = (const float*)d_in[14];
  float* ws  = (float*)d_ws;
  float* out = (float*)d_out;

  k1_prep<<<BB * SS, 256, 0, stream>>>(x, loc, ind, ws);
  k2_forward<<<BB * SS * 5, 256, 0, stream>>>(ws);
  int n3 = BB * CC * 313;
  k3_smix<<<(n3 + 255) / 256, 256, 0, stream>>>(w1s_re, w1s_im, w2s_re, w2s_im, ws);
  k3_cmix<<<(n3 + 255) / 256, 256, 0, stream>>>(w1_re, w1_im, w2_re, w2_im, ws);
  int n3c = BB * CC * SS * 320;
  k3_sexp<<<(n3c + 255) / 256, 256, 0, stream>>>(w1sb_re, w1sb_im, w2sb_re, w2sb_im, ws);
  k4_inverse<<<BB * SS * 8, 256, 0, stream>>>(ws);
  k5_finalize<<<BB * SS * CC, 256, 0, stream>>>(ind, ws, out);
}
